// AttentiveFPModule_33217277067476
// MI455X (gfx1250) — matrix operand order repaired and checked by reading, not compiled, not run on hardware
//
#include <hip/hip_runtime.h>
#include <hip/hip_bf16.h>

// ---------------------------------------------------------------------------
// Problem constants (from reference setup_inputs)
// ---------------------------------------------------------------------------
#define NX 4096      // coarse points
#define NY 16384     // fine (skip) points
#define CC 256       // x channels
#define CS 128       // skip channels
#define CO 256       // output channels
#define CA (CC + CS) // 384 = concat width (GEMM K)
#define KNN 3

typedef __attribute__((ext_vector_type(16))) __bf16        v16bf;
typedef __attribute__((ext_vector_type(8)))  float         v8f;
typedef __attribute__((ext_vector_type(4)))  unsigned int  u32x4;

// ---------------------------------------------------------------------------
// Workspace layout (bytes)
// ---------------------------------------------------------------------------
static constexpr size_t OFF_ATT  = 0;                       // NX f32
static constexpr size_t OFF_IDX  = OFF_ATT  + NX * 4;       // NY*3 i32
static constexpr size_t OFF_S    = OFF_IDX  + NY * 3 * 4;   // NY*3 f32
static constexpr size_t OFF_W1BF = OFF_S    + NY * 3 * 4;   // CA*CO bf16
static constexpr size_t OFF_AIN  = OFF_W1BF + CA * CO * 2;  // NY*CA bf16
static constexpr size_t OFF_H    = OFF_AIN  + (size_t)NY * CA * 2; // NY*CO f32
static constexpr size_t OFF_SUM  = OFF_H    + (size_t)NY * CO * 4; // CO f32
static constexpr size_t OFF_SUM2 = OFF_SUM  + CO * 4;              // CO f32

// ---------------------------------------------------------------------------
// 0) zero the BN accumulators (workspace is poisoned, and must not carry
//    state between calls)
// ---------------------------------------------------------------------------
__global__ void k_zero(float* gsum, float* gsum2) {
  int i = blockIdx.x * blockDim.x + threadIdx.x;
  if (i < CO) { gsum[i] = 0.f; gsum2[i] = 0.f; }
}

// ---------------------------------------------------------------------------
// 1) att = x @ w_att : one wave per row, wave32 shuffle reduction
// ---------------------------------------------------------------------------
__global__ void k_att(const float* __restrict__ x, const float* __restrict__ w,
                      float* __restrict__ att) {
  int tid  = threadIdx.x;
  int row  = blockIdx.x * 8 + (tid >> 5);   // 8 waves per block
  int lane = tid & 31;
  float p = 0.f;
  const float* xr = x + (size_t)row * CC;
#pragma unroll
  for (int j = lane; j < CC; j += 32) p += xr[j] * w[j];
#pragma unroll
  for (int off = 16; off > 0; off >>= 1) p += __shfl_xor(p, off, 32);
  if (lane == 0) att[row] = p;
}

// ---------------------------------------------------------------------------
// 2) convert W1 -> bf16 (row-major [CA][CO])
// ---------------------------------------------------------------------------
__global__ void k_wconv(const float* __restrict__ W1, __bf16* __restrict__ W1bf) {
  int i = blockIdx.x * blockDim.x + threadIdx.x;
  if (i < CA * CO) W1bf[i] = (__bf16)W1[i];
}

// ---------------------------------------------------------------------------
// 3) copy x_skip -> bf16 right half of Ain[:, CC:CA]
// ---------------------------------------------------------------------------
__global__ void k_xskip(const float* __restrict__ xs, __bf16* __restrict__ Ain) {
  int i = blockIdx.x * blockDim.x + threadIdx.x;   // NY*CS threads
  int y = i / CS, c = i % CS;
  Ain[(size_t)y * CA + CC + c] = (__bf16)xs[i];
}

// ---------------------------------------------------------------------------
// 4) KNN + fused neighbor coefficients.
//    Whole coarse cloud lives in LDS (52 KB of the 320 KB WGP pool).
//    Per query: top-3 scan, then s_k = (1/d_k) * softmax(att_k) / sum(1/d)
// ---------------------------------------------------------------------------
__global__ void k_knn(const float* __restrict__ pos, const int* __restrict__ batch,
                      const float* __restrict__ pos_skip,
                      const int* __restrict__ batch_skip,
                      const float* __restrict__ att,
                      int* __restrict__ idx_out, float* __restrict__ s_out) {
  __shared__ float         sp[NX * 3];
  __shared__ unsigned char sb[NX];
  int tid = threadIdx.x;
  for (int i = tid; i < NX * 3; i += blockDim.x) sp[i] = pos[i];
  for (int i = tid; i < NX; i += blockDim.x) sb[i] = (unsigned char)batch[i];
  __syncthreads();

  int y = blockIdx.x * blockDim.x + tid;           // grid covers NY exactly
  float px = pos_skip[y * 3 + 0];
  float py = pos_skip[y * 3 + 1];
  float pz = pos_skip[y * 3 + 2];
  int   by = batch_skip[y];

  float d0 = 1e30f, d1 = 1e30f, d2 = 1e30f;
  int   i0 = 0, i1 = 0, i2 = 0;
  for (int j = 0; j < NX; ++j) {
    float dx = sp[j * 3 + 0] - px;
    float dy = sp[j * 3 + 1] - py;
    float dz = sp[j * 3 + 2] - pz;
    float d  = dx * dx + dy * dy + dz * dz;
    if ((int)sb[j] != by) d = 1e30f;
    if (d < d2) {
      if (d < d1) {
        d2 = d1; i2 = i1;
        if (d < d0) { d1 = d0; i1 = i0; d0 = d; i0 = j; }
        else        { d1 = d;  i1 = j; }
      } else { d2 = d; i2 = j; }
    }
  }

  // weights = 1/clip(d,1e-16); softmax over the 3 attention logits
  float w0 = 1.f / fmaxf(d0, 1e-16f);
  float w1 = 1.f / fmaxf(d1, 1e-16f);
  float w2 = 1.f / fmaxf(d2, 1e-16f);
  float a0 = att[i0], a1 = att[i1], a2 = att[i2];
  float m  = fmaxf(a0, fmaxf(a1, a2));
  float e0 = __expf(a0 - m), e1 = __expf(a1 - m), e2 = __expf(a2 - m);
  float pinv = 1.f / (e0 + e1 + e2);
  float dinv = 1.f / (w0 + w1 + w2);

  idx_out[y * 3 + 0] = i0;
  idx_out[y * 3 + 1] = i1;
  idx_out[y * 3 + 2] = i2;
  s_out[y * 3 + 0] = w0 * e0 * pinv * dinv;
  s_out[y * 3 + 1] = w1 * e1 * pinv * dinv;
  s_out[y * 3 + 2] = w2 * e2 * pinv * dinv;
}

// ---------------------------------------------------------------------------
// 5) gather: Ain[y, 0:CC] = bf16( sum_k s_k * x[idx_k, :] )
// ---------------------------------------------------------------------------
__global__ void k_gather(const float* __restrict__ x, const int* __restrict__ idx3,
                         const float* __restrict__ s3, __bf16* __restrict__ Ain) {
  int y = blockIdx.x;
  int c = threadIdx.x;                              // CC threads
  int i0 = idx3[y * 3 + 0], i1 = idx3[y * 3 + 1], i2 = idx3[y * 3 + 2];
  float s0 = s3[y * 3 + 0], s1 = s3[y * 3 + 1], s2 = s3[y * 3 + 2];
  float v = x[(size_t)i0 * CC + c] * s0
          + x[(size_t)i1 * CC + c] * s1
          + x[(size_t)i2 * CC + c] * s2;
  Ain[(size_t)y * CA + c] = (__bf16)v;
}

// ---------------------------------------------------------------------------
// 6) GEMM h = Ain(bf16) @ W1(bf16) + b1, f32 accumulate via WMMA.
//    Block = 8 waves arranged 4(M) x 2(N): block tile = 64 rows x 256 cols.
//    Each wave: 16 rows x 128 cols = 8 accumulator tiles (64 VGPRs) -- keeps
//    total wave pressure ~110 VGPRs so no scratch spills in the K loop.
//    K loop: 12 x 32 with the 32x256 B slab transposed in LDS.
// ---------------------------------------------------------------------------
__global__ void k_gemm(const __bf16* __restrict__ Ain,
                       const __bf16* __restrict__ W1bf,
                       const float* __restrict__ b1, float* __restrict__ h) {
  __shared__ __align__(16) __bf16 Bs[CO][32];       // [col][k] transposed, 16 KB
  int tid   = threadIdx.x;
  int wave  = tid >> 5;
  int lane  = tid & 31;
  int waveM = wave >> 1;                            // 0..3
  int waveN = wave & 1;                             // 0..1
  int m0    = blockIdx.x * 64 + waveM * 16;
  int n0    = waveN * 128;

  v8f acc[8];
  v8f zero = {0.f, 0.f, 0.f, 0.f, 0.f, 0.f, 0.f, 0.f};
#pragma unroll
  for (int nt = 0; nt < 8; ++nt) acc[nt] = zero;

  // A fragment addressing per ISA 16-bit A layout:
  //   lanes 0-15 : row = lane,    K = {0..7} and {16..23}
  //   lanes 16-31: row = lane-16, K = {8..15} and {24..31}
  int rowA = m0 + (lane & 15);
  int kbA  = (lane < 16) ? 0 : 8;
  const __bf16* arow = Ain + (size_t)rowA * CA;

  int colL = lane & 15;
  int kbB  = (lane < 16) ? 0 : 8;

  for (int kt = 0; kt < 12; ++kt) {
    int k0 = kt * 32;
    __syncthreads();
    // stage B slab W1bf[k0..k0+31][0..255] -> Bs[col][k] (transposed)
#pragma unroll
    for (int q = tid; q < 32 * 32; q += 256) {      // 1024 chunks of 8 bf16
      int k = q >> 5, cb = (q & 31) * 8;
      u32x4 vld = *(const u32x4*)(W1bf + (size_t)(k0 + k) * CO + cb);
      const __bf16* pv = (const __bf16*)&vld;
#pragma unroll
      for (int j = 0; j < 8; ++j) Bs[cb + j][k] = pv[j];
    }
    __syncthreads();

    // prefetch next A K-slab (global_prefetch_b8)
    if (kt + 1 < 12) __builtin_prefetch(arow + k0 + 32, 0, 1);

    // A fragment: two 16B global loads
    v16bf a;
    {
      const __bf16* ap = arow + k0 + kbA;
      *((u32x4*)&a)       = *(const u32x4*)(ap);
      *(((u32x4*)&a) + 1) = *(const u32x4*)(ap + 16);
    }

#pragma unroll
    for (int nt = 0; nt < 8; ++nt) {
      // B fragment: lane<16 -> K {0..7,16..23}, lane>=16 -> K {8..15,24..31}
      v16bf b;
      const __bf16* bp = &Bs[n0 + nt * 16 + colL][kbB];
      *((u32x4*)&b)       = *(const u32x4*)(bp);
      *(((u32x4*)&b) + 1) = *(const u32x4*)(bp + 16);
      acc[nt] = __builtin_amdgcn_wmma_f32_16x16x32_bf16(
          false, a, false, b, (short)0, acc[nt], false, false);
    }
  }

  // C/D layout: lanes 0-15 -> M = m0+r, lanes 16-31 -> M = m0+8+r; N = lane&15
  int rbase = (lane < 16) ? m0 : (m0 + 8);
#pragma unroll
  for (int nt = 0; nt < 8; ++nt) {
    int col = n0 + nt * 16 + colL;
    float bias = b1[col];
#pragma unroll
    for (int r = 0; r < 8; ++r)
      h[(size_t)(rbase + r) * CO + col] = acc[nt][r] + bias;
  }
}

// ---------------------------------------------------------------------------
// 7) BN stats: per-column sum / sumsq, coalesced, atomic reduce
// ---------------------------------------------------------------------------
__global__ void k_bnstats(const float* __restrict__ h, float* __restrict__ gsum,
                          float* __restrict__ gsum2) {
  int c  = threadIdx.x;
  int r0 = blockIdx.x * 128;
  float s = 0.f, s2 = 0.f;
  for (int r = 0; r < 128; ++r) {
    float v = h[(size_t)(r0 + r) * CO + c];
    s += v; s2 += v * v;
  }
  atomicAdd(&gsum[c], s);
  atomicAdd(&gsum2[c], s2);
}

// ---------------------------------------------------------------------------
// 8) BN apply + LeakyReLU -> out
// ---------------------------------------------------------------------------
__global__ void k_bnapply(const float* __restrict__ h, const float* __restrict__ gsum,
                          const float* __restrict__ gsum2,
                          const float* __restrict__ gamma,
                          const float* __restrict__ beta, float* __restrict__ out) {
  int c = threadIdx.x;
  int y = blockIdx.x;
  const float invN = 1.f / (float)NY;
  float mean = gsum[c] * invN;
  float var  = gsum2[c] * invN - mean * mean;
  float inv  = rsqrtf(var + 1e-6f);
  float v = (h[(size_t)y * CO + c] - mean) * inv * gamma[c] + beta[c];
  out[(size_t)y * CO + c] = (v > 0.f) ? v : 0.2f * v;
}

// ---------------------------------------------------------------------------
// 9) tuple tail: pos_skip and batch_skip appended to out
// ---------------------------------------------------------------------------
__global__ void k_tail(const float* __restrict__ pos_skip,
                       const int* __restrict__ batch_skip, float* __restrict__ out) {
  int i = blockIdx.x * blockDim.x + threadIdx.x;
  const size_t base = (size_t)NY * CO;
  if (i < NY * 3) out[base + i] = pos_skip[i];
  if (i < NY)     out[base + NY * 3 + i] = (float)batch_skip[i];
}

// ---------------------------------------------------------------------------
extern "C" void kernel_launch(void* const* d_in, const int* in_sizes, int n_in,
                              void* d_out, int out_size, void* d_ws, size_t ws_size,
                              hipStream_t stream) {
  const float* x          = (const float*)d_in[0];
  const float* pos        = (const float*)d_in[1];
  const int*   batch      = (const int*)d_in[2];
  const float* x_skip     = (const float*)d_in[3];
  const float* pos_skip   = (const float*)d_in[4];
  const int*   batch_skip = (const int*)d_in[5];
  const float* w_att      = (const float*)d_in[6];
  const float* W1         = (const float*)d_in[7];
  const float* b1         = (const float*)d_in[8];
  const float* gamma      = (const float*)d_in[9];
  const float* beta       = (const float*)d_in[10];
  float* out = (float*)d_out;

  char* ws = (char*)d_ws;
  float*  att   = (float*)(ws + OFF_ATT);
  int*    idx3  = (int*)(ws + OFF_IDX);
  float*  s3    = (float*)(ws + OFF_S);
  __bf16* W1bf  = (__bf16*)(ws + OFF_W1BF);
  __bf16* Ain   = (__bf16*)(ws + OFF_AIN);
  float*  hbuf  = (float*)(ws + OFF_H);
  float*  gsum  = (float*)(ws + OFF_SUM);
  float*  gsum2 = (float*)(ws + OFF_SUM2);

  k_zero   <<<1, 256, 0, stream>>>(gsum, gsum2);
  k_att    <<<NX / 8, 256, 0, stream>>>(x, w_att, att);
  k_wconv  <<<(CA * CO + 255) / 256, 256, 0, stream>>>(W1, W1bf);
  k_xskip  <<<(NY * CS) / 256, 256, 0, stream>>>(x_skip, Ain);
  k_knn    <<<NY / 256, 256, 0, stream>>>(pos, batch, pos_skip, batch_skip, att,
                                          idx3, s3);
  k_gather <<<NY, CC, 0, stream>>>(x, idx3, s3, Ain);
  k_gemm   <<<NY / 64, 256, 0, stream>>>(Ain, W1bf, b1, hbuf);
  k_bnstats<<<NY / 128, CO, 0, stream>>>(hbuf, gsum, gsum2);
  k_bnapply<<<NY, CO, 0, stream>>>(hbuf, gsum, gsum2, gamma, beta, out);
  k_tail   <<<(NY * 3 + 255) / 256, 256, 0, stream>>>(pos_skip, batch_skip, out);
}
